// SwinTransformerBlock_42588895707531
// MI455X (gfx1250) — matrix operand order repaired and checked by reading, not compiled, not run on hardware
//
#include <hip/hip_runtime.h>
#include <hip/hip_bf16.h>
#include <math.h>

typedef __bf16 bf16_t;
typedef __attribute__((ext_vector_type(16))) __bf16 v16bf;
typedef __attribute__((ext_vector_type(8)))  __bf16 v8bf;
typedef __attribute__((ext_vector_type(8)))  float  v8f;

#define WMMA_BF16(a,b,c) \
  __builtin_amdgcn_wmma_f32_16x16x32_bf16(false,(a),false,(b),(short)0,(c),false,false)

// ---- problem constants (Swin block: B=16, H=W=128, C=96, M=8, S=4, NH=3) ----
constexpr int CB    = 16;
constexpr int CH    = 128;
constexpr int CC    = 96;     // channels
constexpr int CHD   = 32;     // head dim
constexpr int CNH   = 3;      // heads
constexpr int CN    = 64;     // tokens per window (8x8)
constexpr int CNWIN = 256;    // windows per image (16x16)
constexpr int WTOT  = CB * CNWIN;   // 4096 windows
constexpr int ROWS  = CB * CH * CH; // 262144 token rows
constexpr int C3    = 288;    // 3*C
constexpr int MLPH  = 384;    // 4*C

// ---------------- WMMA operand loaders (CDNA5 16x16x32 bf16 layouts) --------
// A (16x32, MxK): lane L<16 -> row L, K in [kh,kh+8) U [16+kh,16+kh+8), kh=0;
//                 lane L+16 -> row L, kh=8.
__device__ __forceinline__ v16bf load_a_bf16(const bf16_t* base, int row0, int k0, int ld) {
  int lane = threadIdx.x & 31;
  int r    = lane & 15;
  int kh   = (lane >> 4) << 3;
  const bf16_t* p = base + (size_t)(row0 + r) * ld + k0 + kh;
  v8bf lo = *(const v8bf*)p;
  v8bf hi = *(const v8bf*)(p + 16);
  v16bf out;
#pragma unroll
  for (int e = 0; e < 8; ++e) { out[e] = lo[e]; out[e + 8] = hi[e]; }
  return out;
}

// Same A layout but source is fp32 (LDS softmax probabilities), convert inline.
__device__ __forceinline__ v16bf load_a_f32(const float* base, int row0, int k0, int ld) {
  int lane = threadIdx.x & 31;
  int r    = lane & 15;
  int kh   = (lane >> 4) << 3;
  const float* p = base + (row0 + r) * ld + k0 + kh;
  v16bf out;
#pragma unroll
  for (int e = 0; e < 8; ++e) { out[e] = (bf16_t)p[e]; out[e + 8] = (bf16_t)p[e + 16]; }
  return out;
}

// B (32x16, KxN): lane L<16 -> col L, K in [0,8) U [16,24); lane L+16 -> col L, K in [8,16) U [24,32).
// Source is W^T (rows = output columns), so per-lane data is contiguous.
__device__ __forceinline__ v16bf load_b_bf16(const bf16_t* wT, int col0, int k0, int ld) {
  int lane = threadIdx.x & 31;
  int n    = lane & 15;
  int ko   = (lane >> 4) << 3;
  const bf16_t* p = wT + (size_t)(col0 + n) * ld + k0 + ko;
  v8bf lo = *(const v8bf*)p;
  v8bf hi = *(const v8bf*)(p + 16);
  v16bf out;
#pragma unroll
  for (int e = 0; e < 8; ++e) { out[e] = lo[e]; out[e + 8] = hi[e]; }
  return out;
}

// Wave-wide LayerNorm of one 96-wide row (3 values per lane), bf16 result -> dst.
__device__ __forceinline__ void ln_row(const float* __restrict__ src,
                                       const float* __restrict__ w,
                                       const float* __restrict__ bia,
                                       bf16_t* __restrict__ dst, int lane) {
  float v0 = src[lane], v1 = src[lane + 32], v2 = src[lane + 64];
  float s = v0 + v1 + v2, q = v0 * v0 + v1 * v1 + v2 * v2;
#pragma unroll
  for (int o = 16; o > 0; o >>= 1) { s += __shfl_xor(s, o, 32); q += __shfl_xor(q, o, 32); }
  float mu  = s * (1.f / 96.f);
  float inv = rsqrtf(q * (1.f / 96.f) - mu * mu + 1e-5f);
  dst[lane]      = (bf16_t)((v0 - mu) * inv * w[lane]      + bia[lane]);
  dst[lane + 32] = (bf16_t)((v1 - mu) * inv * w[lane + 32] + bia[lane + 32]);
  dst[lane + 64] = (bf16_t)((v2 - mu) * inv * w[lane + 64] + bia[lane + 64]);
}

// ------------------- K0: weights -> bf16 transposed ------------------------
__global__ void k_prep(const float* __restrict__ qkvw, const float* __restrict__ pw,
                       const float* __restrict__ w1,   const float* __restrict__ w2,
                       bf16_t* __restrict__ wqT, bf16_t* __restrict__ wpT,
                       bf16_t* __restrict__ w1T, bf16_t* __restrict__ w2T) {
  int e = blockIdx.x * 256 + threadIdx.x;
  if (e < C3 * CC)      { int o = e / CC,   i = e % CC;   wqT[e] = (bf16_t)qkvw[i * C3 + o];  return; }
  e -= C3 * CC;
  if (e < CC * CC)      { int o = e / CC,   i = e % CC;   wpT[e] = (bf16_t)pw[i * CC + o];    return; }
  e -= CC * CC;
  if (e < MLPH * CC)    { int o = e / CC,   i = e % CC;   w1T[e] = (bf16_t)w1[i * MLPH + o];  return; }
  e -= MLPH * CC;
  if (e < CC * MLPH)    { int o = e / MLPH, i = e % MLPH; w2T[e] = (bf16_t)w2[i * CC + o]; }
}

// ---- K1: fused LN1 + roll(-4,-4) + window gather + QKV GEMM (1 WG/window) --
__global__ void k_qkv(const float* __restrict__ x, const float* __restrict__ n1w,
                      const float* __restrict__ n1b, const bf16_t* __restrict__ wqT,
                      const float* __restrict__ qkvb, bf16_t* __restrict__ qkv) {
  extern __shared__ char smem[];
  bf16_t* Xt = (bf16_t*)smem;                // [64][96] bf16 = 12 KB, LN'd window
  int widx = blockIdx.x;
  int b = widx >> 8, wi = widx & 255, wh = wi >> 4, ww = wi & 15;
  int wv = threadIdx.x >> 5, lane = threadIdx.x & 31;
  int dn = lane & 15, dm = (lane >> 4) << 3;

  // phase 0: LayerNorm1 of the 64 (shift-gathered) tokens into LDS
#pragma unroll
  for (int tt = 0; tt < 8; ++tt) {
    int n = wv * 8 + tt;
    int i = n >> 3, j = n & 7;
    int hs = (wh * 8 + i + 4) & 127;         // roll(-4,-4) source coords
    int ws = (ww * 8 + j + 4) & 127;
    const float* src = x + ((size_t)(b * CH + hs) * CH + ws) * CC;
    ln_row(src, n1w, n1b, Xt + n * CC, lane);
  }
  __syncthreads();

  // phase 1: QKV = Xt @ W_qkv + b, 72 tiles / 8 waves
  bf16_t* Y = qkv + (size_t)widx * CN * C3;
#pragma unroll
  for (int c = 0; c < 9; ++c) {
    int t = wv + c * 8;
    int ti = t / 18, cj = t % 18;
    v8f acc = {};
#pragma unroll
    for (int kk = 0; kk < CC; kk += 32) {
      v16bf a  = load_a_bf16(Xt,  ti * 16, kk, CC);
      v16bf bm = load_b_bf16(wqT, cj * 16, kk, CC);
      acc = WMMA_BF16(a, bm, acc);
    }
    int col = cj * 16 + dn;
    float bv = qkvb[col];
#pragma unroll
    for (int r = 0; r < 8; ++r)
      Y[(size_t)(ti * 16 + dm + r) * C3 + col] = (bf16_t)(acc[r] + bv);
  }
}

// -- K2: fused window attention + proj + un-roll + residual (1 WG / window) --
__global__ void k_attn(const bf16_t* __restrict__ qkv, const bf16_t* __restrict__ wpT,
                       const float* __restrict__ rpb, const float* __restrict__ pb,
                       const float* __restrict__ x, float* __restrict__ x1) {
  extern __shared__ char smem[];
  float*  S  = (float*)smem;                  // [3][64][64] fp32 = 48 KB
  bf16_t* VT = (bf16_t*)(smem + 49152);       // [96][64] bf16   = 12 KB
  bf16_t* O  = (bf16_t*)smem;                 // [64][96] bf16 (overlays S later)

  int widx = blockIdx.x;
  int b = widx >> 8, wi = widx & 255, wh = wi >> 4, ww = wi & 15;
  const bf16_t* QKV = qkv + (size_t)widx * CN * C3;
  int wv = threadIdx.x >> 5, lane = threadIdx.x & 31;
  int dn = lane & 15, dm = (lane >> 4) << 3;

  // prefetch this window's QKV slab (36 KB) -> global_prefetch_b8
  __builtin_prefetch((const char*)QKV + (size_t)threadIdx.x * 144, 0, 3);

  // phase 0: V^T into LDS
  for (int e = threadIdx.x; e < CC * CN; e += 256) {
    int c = e >> 6, n = e & 63;
    VT[c * CN + n] = QKV[(size_t)n * C3 + 2 * CC + c];
  }
  // phase 1: S = Q K^T, one WMMA per 16x16 tile (K = HD = 32)
#pragma unroll
  for (int c = 0; c < 6; ++c) {              // 48 tiles / 8 waves
    int t = wv + c * 8;
    int h = t >> 4, rem = t & 15, ti = rem >> 2, tj = rem & 3;
    v16bf a  = load_a_bf16(QKV, ti * 16, h * CHD, C3);                 // Q rows
    v16bf bm = load_b_bf16(QKV + CC + h * CHD, tj * 16, 0, C3);        // K rows = B cols
    v8f acc = {};
    acc = WMMA_BF16(a, bm, acc);
    float* Sp = S + h * 4096;
#pragma unroll
    for (int r = 0; r < 8; ++r) Sp[(ti * 16 + dm + r) * 64 + tj * 16 + dn] = acc[r];
  }
  __syncthreads();

  // phase 2: softmax with on-the-fly rel-pos bias + shift mask; P kept fp32 in S
  if (threadIdx.x < 192) {
    int h = threadIdx.x / 64, n = threadIdx.x % 64;
    int i1 = n >> 3, j1 = n & 7;
    int hr1 = wh * 8 + i1, wr1 = ww * 8 + j1;
    int lab1 = (hr1 < 120 ? 0 : (hr1 < 124 ? 1 : 2)) * 3 + (wr1 < 120 ? 0 : (wr1 < 124 ? 1 : 2));
    float* row = S + h * 4096 + n * 64;
    const float scale = 0.17677669529663687f;  // 1/sqrt(32)
    float mx = -3.4e38f;
    for (int m = 0; m < 64; ++m) {
      int i2 = m >> 3, j2 = m & 7;
      int hr2 = wh * 8 + i2, wr2 = ww * 8 + j2;
      int lab2 = (hr2 < 120 ? 0 : (hr2 < 124 ? 1 : 2)) * 3 + (wr2 < 120 ? 0 : (wr2 < 124 ? 1 : 2));
      int idx = (i1 - i2 + 7) * 15 + (j1 - j2 + 7);
      float lg = row[m] * scale + rpb[idx * CNH + h] + (lab1 != lab2 ? -100.f : 0.f);
      mx = fmaxf(mx, lg);
    }
    float ssum = 0.f;
    for (int m = 0; m < 64; ++m) {
      int i2 = m >> 3, j2 = m & 7;
      int hr2 = wh * 8 + i2, wr2 = ww * 8 + j2;
      int lab2 = (hr2 < 120 ? 0 : (hr2 < 124 ? 1 : 2)) * 3 + (wr2 < 120 ? 0 : (wr2 < 124 ? 1 : 2));
      int idx = (i1 - i2 + 7) * 15 + (j1 - j2 + 7);
      float lg = row[m] * scale + rpb[idx * CNH + h] + (lab1 != lab2 ? -100.f : 0.f);
      float e = __expf(lg - mx);
      row[m] = e; ssum += e;
    }
    float invs = 1.f / ssum;
    for (int m = 0; m < 64; ++m) row[m] *= invs;
  }
  __syncthreads();

  // phase 3a: O = P V (accumulators held in registers; S still live)
  v8f oacc[3];
#pragma unroll
  for (int c = 0; c < 3; ++c) {              // 24 tiles / 8 waves
    int t = wv + c * 8;
    int h = t >> 3, rem = t & 7, ti = rem >> 1, cj = rem & 1;
    v8f acc = {};
#pragma unroll
    for (int kk = 0; kk < CN; kk += 32) {
      v16bf a  = load_a_f32(S + h * 4096, ti * 16, kk, 64);
      v16bf bm = load_b_bf16(VT, h * CHD + cj * 16, kk, CN);
      acc = WMMA_BF16(a, bm, acc);
    }
    oacc[c] = acc;
  }
  __syncthreads();
  // phase 3b: store O (bf16) over the S region
#pragma unroll
  for (int c = 0; c < 3; ++c) {
    int t = wv + c * 8;
    int h = t >> 3, rem = t & 7, ti = rem >> 1, cj = rem & 1;
    int row0 = ti * 16, col0 = h * CHD + cj * 16;
#pragma unroll
    for (int r = 0; r < 8; ++r)
      O[(row0 + dm + r) * CC + col0 + dn] = (bf16_t)oacc[c][r];
  }
  __syncthreads();

  // phase 4: proj + bias + un-roll + shortcut residual -> x1
#pragma unroll
  for (int c = 0; c < 3; ++c) {              // 24 tiles / 8 waves
    int t = wv + c * 8;
    int ti = t / 6, cj = t % 6;
    v8f acc = {};
#pragma unroll
    for (int kk = 0; kk < CC; kk += 32) {
      v16bf a  = load_a_bf16(O,   ti * 16, kk, CC);
      v16bf bm = load_b_bf16(wpT, cj * 16, kk, CC);
      acc = WMMA_BF16(a, bm, acc);
    }
    int col = cj * 16 + dn;
    float bv = pb[col];
#pragma unroll
    for (int r = 0; r < 8; ++r) {
      int n = ti * 16 + dm + r;
      int i = n >> 3, j = n & 7;
      int ho = (wh * 8 + i + 4) & 127;       // roll(+4,+4) back
      int wo = (ww * 8 + j + 4) & 127;
      size_t g = ((size_t)(b * CH + ho) * CH + wo) * CC + col;
      x1[g] = x[g] + acc[r] + bv;
    }
  }
}

// -- K3: fused LN2 + MLP (GEMM -> GELU -> LDS -> GEMM) + residual, 64 rows/WG --
__global__ void k_mlp(const float* __restrict__ x1, const float* __restrict__ n2w,
                      const float* __restrict__ n2b, const bf16_t* __restrict__ w1T,
                      const float* __restrict__ b1,  const bf16_t* __restrict__ w2T,
                      const float* __restrict__ b2,  float* __restrict__ out) {
  extern __shared__ char smem[];
  bf16_t* Xt = (bf16_t*)smem;                // [64][96]  bf16 = 12 KB (LN2 output)
  bf16_t* Hs = (bf16_t*)(smem + 12288);      // [64][384] bf16 = 48 KB (hidden)
  size_t R0 = (size_t)blockIdx.x * 64;
  int wv = threadIdx.x >> 5, lane = threadIdx.x & 31;
  int dn = lane & 15, dm = (lane >> 4) << 3;

  // phase 0: LayerNorm2 of 64 rows into LDS
#pragma unroll
  for (int tt = 0; tt < 8; ++tt) {
    int rloc = wv * 8 + tt;
    ln_row(x1 + (R0 + rloc) * CC, n2w, n2b, Xt + rloc * CC, lane);
  }
  __syncthreads();

  // phase 1: hidden = GELU(Xt @ W1 + b1), 96 tiles / 8 waves
#pragma unroll
  for (int c = 0; c < 12; ++c) {
    int t = wv + c * 8;
    int ti = t / 24, cj = t % 24;
    v8f acc = {};
#pragma unroll
    for (int kk = 0; kk < CC; kk += 32) {
      v16bf a  = load_a_bf16(Xt,  ti * 16, kk, CC);
      v16bf bm = load_b_bf16(w1T, cj * 16, kk, CC);
      acc = WMMA_BF16(a, bm, acc);
    }
    int col = cj * 16 + dn;
    float bv = b1[col];
#pragma unroll
    for (int r = 0; r < 8; ++r) {
      float v = acc[r] + bv;
      float g = 0.5f * v * (1.f + erff(v * 0.70710678118654752f));  // exact GELU
      Hs[(ti * 16 + dm + r) * MLPH + col] = (bf16_t)g;
    }
  }
  __syncthreads();

  // phase 2: out = x1 + hidden @ W2 + b2, 24 tiles / 8 waves
#pragma unroll
  for (int c = 0; c < 3; ++c) {
    int t = wv + c * 8;
    int ti = t / 6, cj = t % 6;
    v8f acc = {};
#pragma unroll
    for (int kk = 0; kk < MLPH; kk += 32) {
      v16bf a  = load_a_bf16(Hs,  ti * 16, kk, MLPH);
      v16bf bm = load_b_bf16(w2T, cj * 16, kk, MLPH);
      acc = WMMA_BF16(a, bm, acc);
    }
    int col = cj * 16 + dn;
    float bv = b2[col];
#pragma unroll
    for (int r = 0; r < 8; ++r) {
      size_t g = (R0 + ti * 16 + dm + r) * CC + col;
      out[g] = x1[g] + acc[r] + bv;
    }
  }
}

// --------------------------------- launcher ---------------------------------
extern "C" void kernel_launch(void* const* d_in, const int* in_sizes, int n_in,
                              void* d_out, int out_size, void* d_ws, size_t ws_size,
                              hipStream_t stream) {
  const float* x    = (const float*)d_in[0];
  const float* n1w  = (const float*)d_in[1];
  const float* n1b  = (const float*)d_in[2];
  const float* qkvw = (const float*)d_in[3];
  const float* qkvb = (const float*)d_in[4];
  const float* rpb  = (const float*)d_in[5];
  const float* pw   = (const float*)d_in[6];
  const float* pbv  = (const float*)d_in[7];
  const float* n2w  = (const float*)d_in[8];
  const float* n2b  = (const float*)d_in[9];
  const float* w1   = (const float*)d_in[10];
  const float* b1   = (const float*)d_in[11];
  const float* w2   = (const float*)d_in[12];
  const float* b2   = (const float*)d_in[13];
  float* out = (float*)d_out;

  char* ws = (char*)d_ws;
  size_t off = 0;
  bf16_t* qkv  = (bf16_t*)(ws + off); off += (size_t)WTOT * CN * C3 * 2;   // 151 MB
  float*  x1   = (float*) (ws + off); off += (size_t)ROWS * CC * 4;        // 100.7 MB
  bf16_t* wqT  = (bf16_t*)(ws + off); off += (size_t)C3 * CC * 2;
  bf16_t* wpT  = (bf16_t*)(ws + off); off += (size_t)CC * CC * 2;
  bf16_t* w1T  = (bf16_t*)(ws + off); off += (size_t)MLPH * CC * 2;
  bf16_t* w2T  = (bf16_t*)(ws + off); off += (size_t)CC * MLPH * 2;

  k_prep<<<432, 256, 0, stream>>>(qkvw, pw, w1, w2, wqT, wpT, w1T, w2T);
  k_qkv <<<WTOT, 256, 12288, stream>>>(x, n1w, n1b, wqT, qkvb, qkv);
  k_attn<<<WTOT, 256, 61440, stream>>>(qkv, wpT, rpb, pbv, x, x1);
  k_mlp <<<WTOT, 256, 61440, stream>>>(x1, n2w, n2b, w1T, b1, w2T, b2, out);
}
